// MoE_22591527977486
// MI455X (gfx1250) — compile-verified, matrix operand order checked by reading, not run on hardware
//
#include <hip/hip_runtime.h>

// ---------------------------------------------------------------------------
// MoE (B=1024, PHASE=64, GATE=128, K=8, D=512) for MI455X / gfx1250.
//
// Roofline: 12.9 GFLOP, weights ~25 MB fp32 (L2-resident; 192 MB L2).
// Compute-bound -> v_wmma_f32_16x16x32_bf16 (bf16 in, f32 accumulate).
//
// Pipeline:
//   1) gate_kernel : gate MLP + softmax -> coeff[B,K]
//   2) cvt_kernel  : x fp32 -> bf16 activations
//   3) wcvt_kernel : W fp32 [K,i,o] -> bf16 transposed [K,o,i] (x3)
//   4) moe_layer x3: TDM (tensor_load_to_lds) double-buffered weight slabs,
//                    per-expert WMMA GEMMs, coeff-weighted reduction,
//                    mixed bias, ELU. TDM pad gives conflict-free LDS stride.
// ---------------------------------------------------------------------------

typedef __attribute__((ext_vector_type(16))) __bf16 v16bf;
typedef __attribute__((ext_vector_type(8)))  __bf16 v8bf;
typedef __attribute__((ext_vector_type(8)))  float  v8f;
typedef __attribute__((ext_vector_type(4)))  unsigned int v4u;
typedef __attribute__((ext_vector_type(8)))  int v8i;
typedef __attribute__((ext_vector_type(4)))  int v4i;

#define B_ROWS   1024
#define PHASE_N  64
#define GATE_N   128
#define KEXP     8
#define DIM      512
#define KBLK     32     // K-depth per WMMA stage
#define NCHUNK   128    // columns handled per workgroup
#define LDSSTR   40     // row stride (bf16): 32 data + 8 pad -> 80B = 20 banks,
                        // n*20 % 64 distinct for n=0..15 (conflict-free b128)
#define NSTAGE   (KEXP * (DIM / KBLK))   // 128 pipeline stages

#if defined(__has_builtin)
#  if __has_builtin(__builtin_amdgcn_tensor_load_to_lds)
#    define HAS_TDM 1
#  endif
#endif
#ifndef HAS_TDM
#  define HAS_TDM 0
#endif

static __device__ __forceinline__ __bf16 f2bf(float f) {
  unsigned u = __builtin_bit_cast(unsigned, f);
  u += 0x7FFFu + ((u >> 16) & 1u);            // round-to-nearest-even
  unsigned short h = (unsigned short)(u >> 16);
  return __builtin_bit_cast(__bf16, h);
}

static __device__ __forceinline__ float eluf(float x) {
  return x > 0.0f ? x : (__expf(x) - 1.0f);
}

// ---------------------------------------------------------------------------
// Gate MLP + softmax. One 128-thread block per batch row (~52 MFLOP total).
// ---------------------------------------------------------------------------
__global__ __launch_bounds__(128) void gate_kernel(
    const float* __restrict__ phase,
    const float* __restrict__ gw0, const float* __restrict__ gb0,
    const float* __restrict__ gw1, const float* __restrict__ gb1,
    const float* __restrict__ gw2, const float* __restrict__ gb2,
    float* __restrict__ coeff) {
  __shared__ float ph[PHASE_N];
  __shared__ float h0[GATE_N];
  __shared__ float h1[GATE_N];
  __shared__ float lg[KEXP];
  const int r = blockIdx.x;
  const int t = threadIdx.x;

  if (t < PHASE_N) ph[t] = phase[r * PHASE_N + t];
  __syncthreads();
  {
    float s = gb0[t];
    for (int p = 0; p < PHASE_N; ++p) s += ph[p] * gw0[p * GATE_N + t];
    h0[t] = eluf(s);
  }
  __syncthreads();
  {
    float s = gb1[t];
    for (int p = 0; p < GATE_N; ++p) s += h0[p] * gw1[p * GATE_N + t];
    h1[t] = eluf(s);
  }
  __syncthreads();
  if (t < KEXP) {
    float s = gb2[t];
    for (int p = 0; p < GATE_N; ++p) s += h1[p] * gw2[p * KEXP + t];
    lg[t] = s;
  }
  __syncthreads();
  if (t < KEXP) {
    float mx = lg[0];
    for (int i = 1; i < KEXP; ++i) mx = fmaxf(mx, lg[i]);
    float se = 0.0f;
    for (int i = 0; i < KEXP; ++i) se += __expf(lg[i] - mx);
    coeff[r * KEXP + t] = __expf(lg[t] - mx) / se;
  }
}

// ---------------------------------------------------------------------------
// fp32 -> bf16 (input activations)
// ---------------------------------------------------------------------------
__global__ __launch_bounds__(256) void cvt_kernel(const float* __restrict__ x,
                                                  __bf16* __restrict__ y, int n) {
  for (int i = blockIdx.x * blockDim.x + threadIdx.x; i < n;
       i += gridDim.x * blockDim.x)
    y[i] = f2bf(x[i]);
}

// ---------------------------------------------------------------------------
// Weight convert + transpose: W fp32 [K][i][o] -> Wt bf16 [K][o][i].
// 32x32 tiles via LDS; coalesced on both sides. Grid = K*16*16 blocks.
// ---------------------------------------------------------------------------
__global__ __launch_bounds__(256) void wcvt_kernel(const float* __restrict__ W,
                                                   __bf16* __restrict__ Wt) {
  __shared__ float tile[32][33];
  const int bx = blockIdx.x;
  const int k  = bx >> 8;
  const int it = (bx >> 4) & 15;
  const int ot = bx & 15;
  const int t  = threadIdx.x;

  const int oL = t & 31, iL0 = t >> 5;
#pragma unroll
  for (int r = 0; r < 4; ++r) {
    const int iL = iL0 + 8 * r;
    tile[iL][oL] =
        W[((size_t)k * DIM + it * 32 + iL) * DIM + ot * 32 + oL];
  }
  __syncthreads();
  const int iL2 = t & 31, oL0 = t >> 5;
#pragma unroll
  for (int r = 0; r < 4; ++r) {
    const int oL = oL0 + 8 * r;
    Wt[((size_t)k * DIM + ot * 32 + oL) * DIM + it * 32 + iL2] =
        f2bf(tile[iL2][oL]);
  }
}

// ---------------------------------------------------------------------------
// TDM issue: 2-D tile (tile_dim0 = 32 contiguous i, tile_dim1 = 128 rows of o,
// tensor_dim0_stride = 512) from bf16 Wt into LDS, with padding 4 DWORDs after
// every 16 DWORDs -> LDS row stride of 40 bf16 (bank-conflict-free reads).
// ---------------------------------------------------------------------------
#if HAS_TDM
static __device__ __forceinline__ void tdm_issue(unsigned lds_off,
                                                 unsigned long long gaddr) {
  v4u g0;
  g0.x = 1u;                                   // count=1, no gather
  g0.y = lds_off;                              // lds_addr
  g0.z = (unsigned)gaddr;                      // global_addr[31:0]
  g0.w = (unsigned)((gaddr >> 32) & 0x01FFFFFFu) | (2u << 30);  // type=2
  v8i g1;
  g1[0] = (1 << 16)      // data_size = 2 bytes
        | (1 << 20)      // pad_enable
        | (3 << 22)      // pad_interval: 16 DWORDs
        | (3 << 25);     // pad_amount: 4 DWORDs
  g1[1] = (DIM & 0xFFFF) << 16;            // tensor_dim0 lo16 (=512)
  g1[2] = (int)(((KEXP * DIM) & 0xFFFF) << 16);  // dim0 hi=0 | tensor_dim1 lo
  g1[3] = (KBLK << 16);                    // dim1 hi=0 | tile_dim0 = 32
  g1[4] = NCHUNK;                          // tile_dim1 = 128, tile_dim2 = 0
  g1[5] = DIM;                             // tensor_dim0_stride lo32 = 512
  g1[6] = 0;
  g1[7] = 0;
  v4i z4 = {0, 0, 0, 0};                   // 2-D: groups 2/3 unused
  v8i z8 = {0, 0, 0, 0, 0, 0, 0, 0};
  // clang-23 / therock signature: (v4u, v8i, v4i, v4i, v8i, i32 cpol)
  __builtin_amdgcn_tensor_load_to_lds(g0, g1, z4, z4, z8, 0);
}
#endif

// ---------------------------------------------------------------------------
// One MoE layer. Grid: (B/16)*(D/NCHUNK) = 256 WGs of 256 threads (8 wave32).
// Each wave owns one 16x16 output tile: 8 experts x 16 K-blocks = 128 WMMAs.
// Weight slabs stream LDS-ward via TDM, double buffered; DMA of slab s+1
// overlaps WMMA on slab s.
// ---------------------------------------------------------------------------
template <int FINAL>
__global__ __launch_bounds__(256) void moe_layer(
    const __bf16* __restrict__ Ain,   // [B, D] bf16 activations
    const __bf16* __restrict__ Wt,    // [K, D(o), D(i)] bf16 transposed weights
    const float*  __restrict__ bias,  // [K, D] fp32
    const float*  __restrict__ coeff, // [B, K] fp32
    void* __restrict__ outv) {        // bf16 [B,D] or f32 [B,D]
  __shared__ __attribute__((aligned(16))) __bf16 Wlds[2][NCHUNK * LDSSTR];
  __shared__ float cT[16 * KEXP];

  const int tid   = threadIdx.x;
  const int wave  = tid >> 5;
  const int lane  = tid & 31;
  const int hi    = lane >> 4;
  const int lo    = lane & 15;
  const int m0    = (blockIdx.x >> 2) * 16;
  const int ncol0 = (blockIdx.x & 3) * NCHUNK;
  const int n     = ncol0 + wave * 16 + lo;

  if (tid < 16 * KEXP) cT[tid] = coeff[m0 * KEXP + tid];

  const __bf16* arow = Ain + (size_t)(m0 + lo) * DIM;
  const unsigned long long gbase = (unsigned long long)(uintptr_t)Wt;
  const unsigned lds0 = (unsigned)(uintptr_t)(&Wlds[0][0]);
  const unsigned lds1 = (unsigned)(uintptr_t)(&Wlds[1][0]);

  // Global byte address of the weight slab for stage s (expert s>>4, kb block)
  auto slab_addr = [&](int s) -> unsigned long long {
    const int k = s >> 4;
    const int kb = (s & 15) * KBLK;
    return gbase + 2ull * (((size_t)k * DIM + ncol0) * DIM + kb);
  };

#if HAS_TDM
  if (tid < 32) tdm_issue(lds0, slab_addr(0));   // prime the pipeline
#endif

  v8f accF = {0.f, 0.f, 0.f, 0.f, 0.f, 0.f, 0.f, 0.f};
  v8f accE = {0.f, 0.f, 0.f, 0.f, 0.f, 0.f, 0.f, 0.f};

  for (int s = 0; s < NSTAGE; ++s) {
    const int k  = s >> 4;
    const int kb = (s & 15) * KBLK;
    const int buf = s & 1;

#if HAS_TDM
    if (tid < 32) __builtin_amdgcn_s_wait_tensorcnt(0);  // slab s resident
    __syncthreads();  // publish slab s; prior readers of buf^1 are done
    if (s + 1 < NSTAGE && tid < 32)
      tdm_issue(buf ? lds0 : lds1, slab_addr(s + 1));    // overlap next DMA
#else
    (void)lds0; (void)lds1;
    __syncthreads();
    // Fallback: cooperative b128 copy of bf16 slab into padded LDS layout.
    for (int c = tid; c < NCHUNK * (KBLK / 8); c += 256) {
      const int o = c >> 2, part = c & 3;
      *(v8bf*)&Wlds[buf][o * LDSSTR + part * 8] =
          *(const v8bf*)(Wt + ((size_t)k * DIM + ncol0 + o) * DIM + kb +
                         part * 8);
    }
    __syncthreads();
#endif

    // A fragment: two contiguous 16B chunks (K offsets 8*hi, 16+8*hi).
    const __bf16* ap = arow + kb + 8 * hi;
    v8bf a0 = *(const v8bf*)(ap);
    v8bf a1 = *(const v8bf*)(ap + 16);
    v16bf a;
#pragma unroll
    for (int j = 0; j < 8; ++j) { a[j] = a0[j]; a[8 + j] = a1[j]; }

    // B fragment from padded LDS (conflict-free 16B reads).
    const __bf16* bp = &Wlds[buf][(wave * 16 + lo) * LDSSTR + 8 * hi];
    v8bf b0 = *(const v8bf*)(bp);
    v8bf b1 = *(const v8bf*)(bp + 16);
    v16bf b;
#pragma unroll
    for (int j = 0; j < 8; ++j) { b[j] = b0[j]; b[8 + j] = b1[j]; }

    accE = __builtin_amdgcn_wmma_f32_16x16x32_bf16(
        false, a, false, b, (short)0, accE, false, false);

    if ((s & 15) == 15) {  // expert boundary: fold coeff, reset accE
#pragma unroll
      for (int j = 0; j < 8; ++j) {
        const float c = cT[(j + 8 * hi) * KEXP + k];
        accF[j] = fmaf(c, accE[j], accF[j]);
        accE[j] = 0.0f;
      }
    }
  }

  // Epilogue: mixed bias (coeff @ bias), optional ELU, store.
  float bk[KEXP];
#pragma unroll
  for (int kk = 0; kk < KEXP; ++kk) bk[kk] = bias[kk * DIM + n];

#pragma unroll
  for (int j = 0; j < 8; ++j) {
    float mb = 0.0f;
#pragma unroll
    for (int kk = 0; kk < KEXP; ++kk)
      mb = fmaf(cT[(j + 8 * hi) * KEXP + kk], bk[kk], mb);
    float v = accF[j] + mb;
    const int row = m0 + j + 8 * hi;
    if constexpr (FINAL) {
      ((float*)outv)[(size_t)row * DIM + n] = v;
    } else {
      ((__bf16*)outv)[(size_t)row * DIM + n] = f2bf(eluf(v));
    }
  }
}

// ---------------------------------------------------------------------------
extern "C" void kernel_launch(void* const* d_in, const int* in_sizes, int n_in,
                              void* d_out, int out_size, void* d_ws,
                              size_t ws_size, hipStream_t stream) {
  (void)in_sizes; (void)n_in; (void)out_size; (void)ws_size;

  const float* x     = (const float*)d_in[0];
  const float* phase = (const float*)d_in[1];
  const float* gw0   = (const float*)d_in[2];
  const float* gb0   = (const float*)d_in[3];
  const float* gw1   = (const float*)d_in[4];
  const float* gb1   = (const float*)d_in[5];
  const float* gw2   = (const float*)d_in[6];
  const float* gb2   = (const float*)d_in[7];
  const float* W0    = (const float*)d_in[8];
  const float* b0    = (const float*)d_in[9];
  const float* W1    = (const float*)d_in[10];
  const float* b1    = (const float*)d_in[11];
  const float* W2    = (const float*)d_in[12];
  const float* b2    = (const float*)d_in[13];

  // Workspace: coeff 32KB | y0 1MB | y1 1MB | Wt0..2 bf16 (4MB each).
  char*   ws    = (char*)d_ws;
  float*  coeff = (float*)ws;
  __bf16* y0    = (__bf16*)(ws + 32768);
  __bf16* y1    = (__bf16*)(ws + 32768 + (size_t)B_ROWS * DIM * 2);
  __bf16* Wt0   = (__bf16*)(ws + 32768 + 2 * (size_t)B_ROWS * DIM * 2);
  __bf16* Wt1   = Wt0 + (size_t)KEXP * DIM * DIM;
  __bf16* Wt2   = Wt1 + (size_t)KEXP * DIM * DIM;

  gate_kernel<<<B_ROWS, 128, 0, stream>>>(phase, gw0, gb0, gw1, gb1, gw2, gb2,
                                          coeff);
  cvt_kernel<<<512, 256, 0, stream>>>(x, y0, B_ROWS * DIM);

  const dim3 wgrid(KEXP * 16 * 16);  // 2048 blocks of 256
  wcvt_kernel<<<wgrid, 256, 0, stream>>>(W0, Wt0);
  wcvt_kernel<<<wgrid, 256, 0, stream>>>(W1, Wt1);
  wcvt_kernel<<<wgrid, 256, 0, stream>>>(W2, Wt2);

  const dim3 grid((B_ROWS / 16) * (DIM / NCHUNK));  // 256 WGs
  moe_layer<0><<<grid, 256, 0, stream>>>(y0, Wt0, b0, coeff, (void*)y1);
  moe_layer<0><<<grid, 256, 0, stream>>>(y1, Wt1, b1, coeff, (void*)y0);
  moe_layer<1><<<grid, 256, 0, stream>>>(y0, Wt2, b2, coeff, d_out);
}
